// WindowAttention_67997922230605
// MI455X (gfx1250) — matrix operand order repaired and checked by reading, not compiled, not run on hardware
//
#include <hip/hip_runtime.h>

typedef __attribute__((ext_vector_type(2)))  float  v2f;
typedef __attribute__((ext_vector_type(8)))  float  v8f;
typedef __attribute__((ext_vector_type(16))) __bf16 v16bf;
typedef __attribute__((ext_vector_type(8)))  __bf16 v8bf;
typedef __attribute__((ext_vector_type(4)))  __bf16 v4bf;

#define D_MODEL 256
#define NHEAD   8
#define HDIM    32
#define TOK     64          // padded tokens per window
#define NW      2048
#define S_BI    264         // stride (bf16 elems) of [64][264] bf16 LDS planes
#define S_H     33          // stride (floats) of per-head Q/K/V LDS tiles [64][33]
#define S_AT    66          // stride (floats) of per-wave attn scratch [16][66]

// ---- LDS byte offsets (regions reused over time; see barriers) ----
#define IN_HI_OFF 0                      // qk_in hi plane [64][264] bf16 (33792 B)
#define IN_LO_OFF 33792                  // qk_in lo plane
#define FT_HI_OFF 67584                  // feat hi plane
#define FT_LO_OFF 101376                 // feat lo plane
#define VH_OFF    135168                 // per-head V f32 [8][64][33] (67584 B, dedicated)
#define ATTN_OFF  202752                 // per-wave attn scratch f32 [8][16][66] (33792 B)
#define SMEM_BYTES 236544
// Aliases (time-multiplexed):
//   qh (f32 [8][64][33]) = FT_HI_OFF   (after all waves finished reading feat planes)
//   kh (f32 [8][64][33]) = IN_HI_OFF   (after all waves finished reading qk_in planes)
//   sOut hi/lo bf16      = IN_HI/IN_LO (after all waves finished attention)

static __device__ __forceinline__ v8f wmma4(v2f a, v2f b, v8f c) {
  return __builtin_amdgcn_wmma_f32_16x16x4_f32(false, a, false, b, (short)0, c,
                                               false, false);
}
static __device__ __forceinline__ v8f wmma_bf(v16bf a, v16bf b, v8f c) {
  return __builtin_amdgcn_wmma_f32_16x16x32_bf16(false, a, false, b, (short)0, c,
                                                 false, false);
}
static __device__ __forceinline__ v8f vzero8() {
  v8f z;
#pragma unroll
  for (int i = 0; i < 8; ++i) z[i] = 0.0f;
  return z;
}
static __device__ __forceinline__ v16bf cat8(v8bf a, v8bf b) {
  v16bf r;
#pragma unroll
  for (int i = 0; i < 8; ++i) { r[i] = a[i]; r[8 + i] = b[i]; }
  return r;
}

// A operand (16x32 bf16): lane half 0 needs K {0..7,16..23}, half 1 {8..15,24..31};
// src is row-major bf16 with stride S_BI -> two contiguous 16B LDS loads.
static __device__ __forceinline__ v16bf load_a_bf(const __bf16* src, int row,
                                                  int k0, int half) {
  const __bf16* p = src + row * S_BI + k0 + half * 8;
  return cat8(*(const v8bf*)p, *(const v8bf*)(p + 16));
}
// B operand (32x16 bf16): B[k][n] = W[n][k]; lane half 0 needs K {0..7,16..23}, half 1 {8..15,24..31};
// W row-major bf16 [n][256] -> two contiguous 16B global loads.
static __device__ __forceinline__ v16bf load_b_bf(const __bf16* W, int n,
                                                  int k0, int half) {
  const __bf16* p = W + n * D_MODEL + k0 + half * 8;
  return cat8(*(const v8bf*)p, *(const v8bf*)(p + 16));
}

// 3-term bf16 emulation of fp32 GEMM: [64 x 32] = A([64][256]) @ W^T,
// output cols = W rows [ncol0, ncol0+32). Accumulates into acc[4][2] (f32).
static __device__ __forceinline__ void proj_bf3(
    const __bf16* __restrict__ ahi, const __bf16* __restrict__ alo,
    const __bf16* __restrict__ whi, const __bf16* __restrict__ wlo,
    int ncol0, v8f acc[4][2], int lane) {
  const int ln = lane & 15;
  const int half = lane >> 4;
  for (int k0 = 0; k0 < D_MODEL; k0 += 32) {
    v16bf Ah[4], Al[4], Bh[2], Bl[2];
#pragma unroll
    for (int mt = 0; mt < 4; ++mt) {
      Ah[mt] = load_a_bf(ahi, mt * 16 + ln, k0, half);
      Al[mt] = load_a_bf(alo, mt * 16 + ln, k0, half);
    }
#pragma unroll
    for (int nt = 0; nt < 2; ++nt) {
      Bh[nt] = load_b_bf(whi, ncol0 + nt * 16 + ln, k0, half);
      Bl[nt] = load_b_bf(wlo, ncol0 + nt * 16 + ln, k0, half);
    }
#pragma unroll
    for (int mt = 0; mt < 4; ++mt)
#pragma unroll
      for (int nt = 0; nt < 2; ++nt) {
        acc[mt][nt] = wmma_bf(Ah[mt], Bh[nt], acc[mt][nt]);
        acc[mt][nt] = wmma_bf(Ah[mt], Bl[nt], acc[mt][nt]);
        acc[mt][nt] = wmma_bf(Al[mt], Bh[nt], acc[mt][nt]);
      }
  }
}

// Store C-layout tiles (bias-added, scaled) into row-major f32 [64][S_H] LDS.
static __device__ __forceinline__ void store_tiles_h(float* __restrict__ dst,
                                                     const v8f acc[4][2],
                                                     const float* __restrict__ bias,
                                                     int bcol0, float scale,
                                                     int lane) {
  const int ln = lane & 15;
  const int halfr = (lane >> 4) * 8;
#pragma unroll
  for (int mt = 0; mt < 4; ++mt)
#pragma unroll
    for (int nt = 0; nt < 2; ++nt) {
      float bb = bias[bcol0 + nt * 16 + ln];
#pragma unroll
      for (int r = 0; r < 8; ++r)
        dst[(mt * 16 + halfr + r) * S_H + nt * 16 + ln] = (acc[mt][nt][r] + bb) * scale;
    }
}

static __device__ __forceinline__ void split4(float4 v, v4bf* hi, v4bf* lo) {
  float x[4] = {v.x, v.y, v.z, v.w};
#pragma unroll
  for (int i = 0; i < 4; ++i) {
    __bf16 h = (__bf16)x[i];
    (*hi)[i] = h;
    (*lo)[i] = (__bf16)(x[i] - (float)h);
  }
}

__global__ void wa_init_map(int* __restrict__ map, int P) {
  int i = blockIdx.x * 256 + threadIdx.x;
  if (i < P) map[i] = -1;
}

__global__ void wa_build_map(const int* __restrict__ wi, const int* __restrict__ si,
                             int* __restrict__ map, int N) {
  int i = blockIdx.x * 256 + threadIdx.x;
  if (i < N) map[wi[i] * TOK + si[i]] = i;
}

// Split all weights (Wq,Wk,Wv,Wo concatenated) into bf16 hi/lo planes.
__global__ void wa_cvt_w(const float* __restrict__ wio, const float* __restrict__ wo,
                         __bf16* __restrict__ whi, __bf16* __restrict__ wlo) {
  int i = blockIdx.x * 256 + threadIdx.x;  // 0 .. 4*65536-1
  float x = (i < 3 * D_MODEL * D_MODEL) ? wio[i] : wo[i - 3 * D_MODEL * D_MODEL];
  __bf16 h = (__bf16)x;
  whi[i] = h;
  wlo[i] = (__bf16)(x - (float)h);
}

__global__ __launch_bounds__(256, 1)
void wa_attn(const float* __restrict__ vf, const float* __restrict__ pe,
             const float* __restrict__ bio, const float* __restrict__ bo,
             const __bf16* __restrict__ whi, const __bf16* __restrict__ wlo,
             const int* __restrict__ map, float* __restrict__ out) {
  __shared__ __align__(16) unsigned char smem[SMEM_BYTES];
  __shared__ float smask[TOK];
  __shared__ int smap[TOK];

  const int tid = threadIdx.x;
  const int w = blockIdx.x;
  const int lane = tid & 31;
  const int h = tid >> 5;  // wave id == head id

  __bf16* in_hi = (__bf16*)(smem + IN_HI_OFF);
  __bf16* in_lo = (__bf16*)(smem + IN_LO_OFF);
  __bf16* ft_hi = (__bf16*)(smem + FT_HI_OFF);
  __bf16* ft_lo = (__bf16*)(smem + FT_LO_OFF);

  // ---- Phase 0: scatter voxels into padded window, split to bf16 hi/lo ----
  {
    int row = tid >> 2;  // 0..63, 4 threads per row
    int sub = tid & 3;
    int mp = map[w * TOK + row];
    if (sub == 0) {
      smask[row] = (mp >= 0) ? 0.0f : -__builtin_inff();
      smap[row] = mp;
    }
    __bf16* ih = in_hi + row * S_BI;
    __bf16* il = in_lo + row * S_BI;
    __bf16* fh = ft_hi + row * S_BI;
    __bf16* fl = ft_lo + row * S_BI;
    if (mp >= 0) {
      const float4* f4 = (const float4*)(vf + (long)mp * D_MODEL);
      const float4* p4 = (const float4*)(pe + (long)mp * D_MODEL);
      for (int c = sub; c < 64; c += 4) {
        float4 fv = f4[c], pv = p4[c];
        float4 s;
        s.x = fv.x + pv.x; s.y = fv.y + pv.y; s.z = fv.z + pv.z; s.w = fv.w + pv.w;
        v4bf hi, lo;
        split4(fv, &hi, &lo);
        *(v4bf*)(fh + c * 4) = hi; *(v4bf*)(fl + c * 4) = lo;
        split4(s, &hi, &lo);
        *(v4bf*)(ih + c * 4) = hi; *(v4bf*)(il + c * 4) = lo;
      }
    } else {
      v4bf z = {(__bf16)0.f, (__bf16)0.f, (__bf16)0.f, (__bf16)0.f};
      v4bf o = {(__bf16)1.f, (__bf16)1.f, (__bf16)1.f, (__bf16)1.f};
      for (int c = sub; c < 64; c += 4) {
        *(v4bf*)(fh + c * 4) = z; *(v4bf*)(fl + c * 4) = z;
        *(v4bf*)(ih + c * 4) = o; *(v4bf*)(il + c * 4) = z;  // feat 0 + pos 1
      }
    }
  }
  __syncthreads();  // B1

  const float inv_sqrt_hd = 0.17677669529663688f;  // 1/sqrt(32)
  float* vh = (float*)(smem + VH_OFF) + h * TOK * S_H;    // dedicated
  float* qh = (float*)(smem + FT_HI_OFF) + h * TOK * S_H; // reuses feat planes
  float* kh = (float*)(smem + IN_HI_OFF) + h * TOK * S_H; // reuses qk_in planes
  float* at = (float*)(smem + ATTN_OFF) + h * 16 * S_AT;

  const __bf16* whi_q = whi;
  const __bf16* whi_k = whi + D_MODEL * D_MODEL;
  const __bf16* whi_v = whi + 2 * D_MODEL * D_MODEL;
  const __bf16* whi_o = whi + 3 * D_MODEL * D_MODEL;
  const __bf16* wlo_q = wlo;
  const __bf16* wlo_k = wlo + D_MODEL * D_MODEL;
  const __bf16* wlo_v = wlo + 2 * D_MODEL * D_MODEL;
  const __bf16* wlo_o = wlo + 3 * D_MODEL * D_MODEL;

  // ---- Phase 1: V = feat @ Wv^T + bv ----
  {
    v8f acc[4][2];
#pragma unroll
    for (int mt = 0; mt < 4; ++mt)
#pragma unroll
      for (int nt = 0; nt < 2; ++nt) acc[mt][nt] = vzero8();
    proj_bf3(ft_hi, ft_lo, whi_v, wlo_v, h * HDIM, acc, lane);
    __syncthreads();  // B2: all waves done reading feat planes -> qh region free
    store_tiles_h(vh, acc, bio + 2 * D_MODEL, h * HDIM, 1.0f, lane);
  }

  // ---- Phase 2: Q = (qk_in @ Wq^T + bq) * scale ----
  {
    v8f acc[4][2];
#pragma unroll
    for (int mt = 0; mt < 4; ++mt)
#pragma unroll
      for (int nt = 0; nt < 2; ++nt) acc[mt][nt] = vzero8();
    proj_bf3(in_hi, in_lo, whi_q, wlo_q, h * HDIM, acc, lane);
    store_tiles_h(qh, acc, bio, h * HDIM, inv_sqrt_hd, lane);
  }

  // ---- Phase 3: K = qk_in @ Wk^T + bk (compute, barrier, overwrite IN) ----
  {
    v8f acc[4][2];
#pragma unroll
    for (int mt = 0; mt < 4; ++mt)
#pragma unroll
      for (int nt = 0; nt < 2; ++nt) acc[mt][nt] = vzero8();
    proj_bf3(in_hi, in_lo, whi_k, wlo_k, h * HDIM, acc, lane);
    __syncthreads();  // B3: all waves done reading qk_in planes -> kh region free
    store_tiles_h(kh, acc, bio + D_MODEL, h * HDIM, 1.0f, lane);
  }

  // ---- Phase 4: attention per 16-query M-tile (exact f32 WMMA) ----
  const int ln = lane & 15;
  const int koff = (lane >> 4) << 1;
  const int halfr = (lane >> 4) * 8;
  v8f oreg[4][2];
#pragma unroll
  for (int mt = 0; mt < 4; ++mt)
#pragma unroll
    for (int nt = 0; nt < 2; ++nt) oreg[mt][nt] = vzero8();

#pragma unroll
  for (int mt = 0; mt < 4; ++mt) {
    v8f s[4];
#pragma unroll
    for (int j = 0; j < 4; ++j) s[j] = vzero8();

    // scores = Q_tile @ K^T  (K-dim = 32)
#pragma unroll
    for (int k0 = 0; k0 < HDIM; k0 += 4) {
      const float* ap = qh + (mt * 16 + ln) * S_H + k0 + koff;
      v2f a; a.x = ap[0]; a.y = ap[1];
#pragma unroll
      for (int j = 0; j < 4; ++j) {
        const float* bp = kh + (j * 16 + ln) * S_H + k0 + koff;  // B[k][n]=K[n][k]
        v2f b; b.x = bp[0]; b.y = bp[1];
        s[j] = wmma4(a, b, s[j]);
      }
    }
    // key-padding mask
#pragma unroll
    for (int j = 0; j < 4; ++j) {
      float mv = smask[j * 16 + ln];
#pragma unroll
      for (int r = 0; r < 8; ++r) s[j][r] += mv;
    }
    // softmax over 64 keys (width-16 half-wave reductions)
#pragma unroll
    for (int r = 0; r < 8; ++r) {
      float m = fmaxf(fmaxf(s[0][r], s[1][r]), fmaxf(s[2][r], s[3][r]));
      for (int off = 8; off; off >>= 1) m = fmaxf(m, __shfl_xor(m, off, 16));
      float sum = 0.0f;
#pragma unroll
      for (int j = 0; j < 4; ++j) {
        float e = __expf(s[j][r] - m);
        s[j][r] = e;
        sum += e;
      }
      for (int off = 8; off; off >>= 1) sum += __shfl_xor(sum, off, 16);
      float inv = 1.0f / sum;
#pragma unroll
      for (int j = 0; j < 4; ++j) s[j][r] *= inv;
    }
    // stage attn to LDS so it can be re-read in A-layout
#pragma unroll
    for (int j = 0; j < 4; ++j)
#pragma unroll
      for (int r = 0; r < 8; ++r) at[(halfr + r) * S_AT + j * 16 + ln] = s[j][r];

    // out_tile += attn(16x64) @ V(64x32)
    for (int k0 = 0; k0 < TOK; k0 += 4) {
      const float* ap = at + ln * S_AT + k0 + koff;
      v2f a; a.x = ap[0]; a.y = ap[1];
#pragma unroll
      for (int nt = 0; nt < 2; ++nt) {
        v2f b;
        b.x = vh[(k0 + koff) * S_H + nt * 16 + ln];
        b.y = vh[(k0 + koff + 1) * S_H + nt * 16 + ln];
        oreg[mt][nt] = wmma4(a, b, oreg[mt][nt]);
      }
    }
  }
  __syncthreads();  // B4: everyone done reading kh (IN region) -> sOut free

  // ---- Phase 5: concat head outputs, split to bf16 hi/lo planes ----
  __bf16* so_hi = (__bf16*)(smem + IN_HI_OFF);
  __bf16* so_lo = (__bf16*)(smem + IN_LO_OFF);
#pragma unroll
  for (int mt = 0; mt < 4; ++mt)
#pragma unroll
    for (int nt = 0; nt < 2; ++nt)
#pragma unroll
      for (int r = 0; r < 8; ++r) {
        float vv = oreg[mt][nt][r];
        __bf16 hh = (__bf16)vv;
        int idx = (mt * 16 + halfr + r) * S_BI + h * HDIM + nt * 16 + ln;
        so_hi[idx] = hh;
        so_lo[idx] = (__bf16)(vv - (float)hh);
      }
  __syncthreads();  // B5

  // ---- Phase 6: out = attnout @ Wo^T + bo; scatter valid rows to global ----
  {
    v8f acc[4][2];
#pragma unroll
    for (int mt = 0; mt < 4; ++mt)
#pragma unroll
      for (int nt = 0; nt < 2; ++nt) acc[mt][nt] = vzero8();
    proj_bf3(so_hi, so_lo, whi_o, wlo_o, h * HDIM, acc, lane);
#pragma unroll
    for (int mt = 0; mt < 4; ++mt)
#pragma unroll
      for (int nt = 0; nt < 2; ++nt) {
        float bb = bo[h * HDIM + nt * 16 + ln];
#pragma unroll
        for (int r = 0; r < 8; ++r) {
          int row = mt * 16 + halfr + r;
          int mp = smap[row];
          if (mp >= 0)
            out[(long)mp * D_MODEL + h * HDIM + nt * 16 + ln] = acc[mt][nt][r] + bb;
        }
      }
  }
}

extern "C" void kernel_launch(void* const* d_in, const int* in_sizes, int n_in,
                              void* d_out, int out_size, void* d_ws, size_t ws_size,
                              hipStream_t stream) {
  const float* voxel_feat = (const float*)d_in[0];
  const float* pos_embed  = (const float*)d_in[1];
  const float* in_proj_w  = (const float*)d_in[2];
  const float* in_proj_b  = (const float*)d_in[3];
  const float* out_proj_w = (const float*)d_in[4];
  const float* out_proj_b = (const float*)d_in[5];
  const int*   window_idx = (const int*)d_in[6];
  const int*   slot_idx   = (const int*)d_in[7];

  const int N = in_sizes[0] / D_MODEL;  // 98304
  const int P = NW * TOK;               // 131072
  const int NWEL = 4 * D_MODEL * D_MODEL;  // 262144 weight elements (q,k,v,o)

  int*    map = (int*)d_ws;                                   // 512 KB
  __bf16* whi = (__bf16*)((char*)d_ws + (size_t)P * 4);       // 512 KB
  __bf16* wlo = whi + NWEL;                                   // 512 KB

  wa_init_map<<<(P + 255) / 256, 256, 0, stream>>>(map, P);
  wa_build_map<<<(N + 255) / 256, 256, 0, stream>>>(window_idx, slot_idx, map, N);
  wa_cvt_w<<<NWEL / 256, 256, 0, stream>>>(in_proj_w, out_proj_w, whi, wlo);
  wa_attn<<<NW, 256, 0, stream>>>(voxel_feat, pos_embed, in_proj_b, out_proj_b,
                                  whi, wlo, map, (float*)d_out);
}
